// LSTMCell_21354577396155
// MI455X (gfx1250) — matrix operand order repaired and checked by reading, not compiled, not run on hardware
//
#include <hip/hip_runtime.h>

#define B_DIM 8192
#define I_DIM 2048
#define H_DIM 2048

typedef __bf16 bf16_t;
typedef __attribute__((ext_vector_type(16))) __bf16 v16bf;
typedef __attribute__((ext_vector_type(8)))  float  v8f;
typedef int v4i __attribute__((__vector_size__(4 * sizeof(int))));

#if __has_builtin(__builtin_amdgcn_global_load_async_to_lds_b128) && \
    __has_builtin(__builtin_amdgcn_s_wait_asynccnt)
#define HAVE_ASYNC_LDS 1
#else
#define HAVE_ASYNC_LDS 0
#endif

#if HAVE_ASYNC_LDS
// Builtin signature (from compiler diagnostic): param0 = v4i* in global/device
// AS, param1 = LDS side, params 2/3 = imm offset / cpol. Numeric target-AS
// pointers are accepted for the language-AS params; go through inttoptr.
#define AS_GLOBAL(p) ((__attribute__((address_space(1))) v4i*)(size_t)(p))
#define AS_LDS(p)    ((__attribute__((address_space(3))) v4i*)(size_t)(p))
#endif

// ---------------------------------------------------------------------------
// Hardware transcendentals: gfx1250 has V_TANH_F32.
// ---------------------------------------------------------------------------
__device__ __forceinline__ float fast_tanh(float x) {
#if __has_builtin(__builtin_amdgcn_tanhf)
    return __builtin_amdgcn_tanhf(x);
#elif __has_builtin(__builtin_amdgcn_tanh_f32)
    return __builtin_amdgcn_tanh_f32(x);
#else
    return tanhf(x);
#endif
}
__device__ __forceinline__ float fast_sigmoid(float x) {
    return 0.5f * fast_tanh(0.5f * x) + 0.5f;   // sigmoid(x) = (tanh(x/2)+1)/2
}

// ---------------------------------------------------------------------------
// f32 -> bf16 conversion, 8 elements / thread (b128 in/out)
// ---------------------------------------------------------------------------
__global__ void cvt_f32_bf16_kernel(const float* __restrict__ in,
                                    bf16_t* __restrict__ out, int n) {
    int i = (blockIdx.x * blockDim.x + threadIdx.x) * 8;
    const int stride = gridDim.x * blockDim.x * 8;
    for (; i < n; i += stride) {
        const float4 a = *(const float4*)(in + i);
        const float4 b = *(const float4*)(in + i + 4);
        union { bf16_t h[8]; float4 v; } u;
        u.h[0] = (bf16_t)a.x; u.h[1] = (bf16_t)a.y;
        u.h[2] = (bf16_t)a.z; u.h[3] = (bf16_t)a.w;
        u.h[4] = (bf16_t)b.x; u.h[5] = (bf16_t)b.y;
        u.h[6] = (bf16_t)b.z; u.h[7] = (bf16_t)b.w;
        *(float4*)(out + i) = u.v;
    }
}

// ---------------------------------------------------------------------------
// LDS fragment loaders (padded row stride = 40 bf16 = 80 B).
// A (16x32 bf16, ISA 7.12.2): lane m(0..15): K=0..7,16..23; lane m+16: K=8..15,24..31
// B (32x16 bf16):             lane n(0..15): col n, K=0..7,16..23; lane n+16: K=8..15,24..31
// ---------------------------------------------------------------------------
__device__ __forceinline__ v16bf frag_a(const bf16_t* base, int lane) {
    const int row = lane & 15;
    const int kh  = (lane >> 4) * 8;
    v16bf r;
    ((float4*)&r)[0] = *(const float4*)(base + row * 40 + kh);
    ((float4*)&r)[1] = *(const float4*)(base + row * 40 + kh + 16);
    return r;
}
__device__ __forceinline__ v16bf frag_b(const bf16_t* base, int lane) {
    const int col = lane & 15;
    const int kb  = (lane >> 4) * 8;
    v16bf r;
    ((float4*)&r)[0] = *(const float4*)(base + col * 40 + kb);
    ((float4*)&r)[1] = *(const float4*)(base + col * 40 + kb + 16);
    return r;
}

// ---------------------------------------------------------------------------
// Fused LSTM cell.
// Block tile: 128 (batch) x 64 (hidden), all 4 gates. 256 threads = 8 waves.
// Wave (wm = M half of 64 rows, wn = N slice of 16 cols):
//   16 WMMAs (4 M-tiles x 4 gates) per 32-deep k-step.
// K = 4096 total (phase 0: x @ W_x^T, phase 1: h_prev @ W_h^T), 128 steps.
// Double-buffered LDS, async global->LDS staging (ASYNCcnt).
// ---------------------------------------------------------------------------
__global__ __launch_bounds__(256)
void lstm_cell_wmma_kernel(const bf16_t* __restrict__ x_bf,
                           const bf16_t* __restrict__ h_bf,
                           const bf16_t* __restrict__ wx_bf,
                           const bf16_t* __restrict__ wh_bf,
                           const float*  __restrict__ c_prev,
                           const float*  __restrict__ b_x,
                           const float*  __restrict__ b_h,
                           float* __restrict__ c_out,
                           float* __restrict__ h_out) {
    __shared__ __align__(16) bf16_t As[2][128][40];     // 20 KB
    __shared__ __align__(16) bf16_t Bs[2][4][64][40];   // 40 KB

    const int tid  = threadIdx.x;
    const int lane = tid & 31;
    const int wave = tid >> 5;
    const int wm   = wave >> 2;   // 0..1 : which 64-row half
    const int wn   = wave & 3;    // 0..3 : which 16-col slice

    const int nBase = blockIdx.x * 64;    // hidden base
    const int bBase = blockIdx.y * 128;   // batch base

    // staging coordinates (per thread, constant across steps)
    const int ar = tid >> 1;         // A row 0..127
    const int ac = (tid & 1) * 16;   // A col block {0,16}: two b128 each
    const int bg = tid >> 6;         // B gate 0..3
    const int bh = tid & 63;         // B row (hidden) 0..63: four b128 each

    v8f acc[4][4];
#pragma unroll
    for (int g = 0; g < 4; ++g)
#pragma unroll
        for (int mt = 0; mt < 4; ++mt)
            acc[g][mt] = (v8f){0.f, 0.f, 0.f, 0.f, 0.f, 0.f, 0.f, 0.f};

    // stage step s's tiles into LDS buffer `buf` (6 b128 transfers / thread)
    auto stage = [&](int buf, int s) {
        const int phase = s >> 6;
        const int k0    = (s & 63) * 32;
        const bf16_t* Ag = (phase ? h_bf : x_bf)
                         + (size_t)(bBase + ar) * 2048 + k0 + ac;
        const bf16_t* Wg = (phase ? wh_bf : wx_bf)
                         + ((size_t)bg * H_DIM + (nBase + bh)) * 2048 + k0;
#if HAVE_ASYNC_LDS
        __builtin_amdgcn_global_load_async_to_lds_b128(
            AS_GLOBAL(Ag),      AS_LDS(&As[buf][ar][ac]),      0, 0);
        __builtin_amdgcn_global_load_async_to_lds_b128(
            AS_GLOBAL(Ag + 8),  AS_LDS(&As[buf][ar][ac + 8]),  0, 0);
        __builtin_amdgcn_global_load_async_to_lds_b128(
            AS_GLOBAL(Wg),      AS_LDS(&Bs[buf][bg][bh][0]),   0, 0);
        __builtin_amdgcn_global_load_async_to_lds_b128(
            AS_GLOBAL(Wg + 8),  AS_LDS(&Bs[buf][bg][bh][8]),   0, 0);
        __builtin_amdgcn_global_load_async_to_lds_b128(
            AS_GLOBAL(Wg + 16), AS_LDS(&Bs[buf][bg][bh][16]),  0, 0);
        __builtin_amdgcn_global_load_async_to_lds_b128(
            AS_GLOBAL(Wg + 24), AS_LDS(&Bs[buf][bg][bh][24]),  0, 0);
#else
        *(float4*)(&As[buf][ar][ac])     = *(const float4*)(Ag);
        *(float4*)(&As[buf][ar][ac + 8]) = *(const float4*)(Ag + 8);
        *(float4*)(&Bs[buf][bg][bh][0])  = *(const float4*)(Wg);
        *(float4*)(&Bs[buf][bg][bh][8])  = *(const float4*)(Wg + 8);
        *(float4*)(&Bs[buf][bg][bh][16]) = *(const float4*)(Wg + 16);
        *(float4*)(&Bs[buf][bg][bh][24]) = *(const float4*)(Wg + 24);
#endif
    };

    stage(0, 0);   // prologue

#pragma unroll 1
    for (int s = 0; s < 128; ++s) {
        const int buf = s & 1;
        if (s + 1 < 128) {
            stage(buf ^ 1, s + 1);        // prefetch next tile while computing
#if HAVE_ASYNC_LDS
            __builtin_amdgcn_s_wait_asynccnt(6);   // current tile's 6 done
#endif
        } else {
#if HAVE_ASYNC_LDS
            __builtin_amdgcn_s_wait_asynccnt(0);
#endif
        }
        __syncthreads();                  // tile `buf` visible to all waves

        v16bf a[4];
#pragma unroll
        for (int mt = 0; mt < 4; ++mt)
            a[mt] = frag_a(&As[buf][wm * 64 + mt * 16][0], lane);
        v16bf bb[4];
#pragma unroll
        for (int g = 0; g < 4; ++g)
            bb[g] = frag_b(&Bs[buf][g][wn * 16][0], lane);

#pragma unroll
        for (int g = 0; g < 4; ++g)
#pragma unroll
            for (int mt = 0; mt < 4; ++mt)
                acc[g][mt] = __builtin_amdgcn_wmma_f32_16x16x32_bf16(
                    false, a[mt], false, bb[g], (short)0, acc[g][mt],
                    false, false);

        __syncthreads();                  // safe to overwrite buffer `buf`
    }

    // ---------------- epilogue: biases + gate nonlinearities ----------------
    // C/D layout: VGPR j -> M = j (lanes 0-15) or j+8 (lanes 16-31); N = lane&15
    const int nl = lane & 15;
    const int mh = (lane >> 4) * 8;
    const int n  = nBase + wn * 16 + nl;

    float bias[4];
#pragma unroll
    for (int g = 0; g < 4; ++g)
        bias[g] = b_x[g * H_DIM + n] + b_h[g * H_DIM + n];

#pragma unroll
    for (int mt = 0; mt < 4; ++mt) {
#pragma unroll
        for (int j = 0; j < 8; ++j) {
            const int row = bBase + wm * 64 + mt * 16 + mh + j;
            const float gf = acc[0][mt][j] + bias[0];
            const float gi = acc[1][mt][j] + bias[1];
            const float gc = acc[2][mt][j] + bias[2];
            const float go = acc[3][mt][j] + bias[3];

            const float f  = fast_sigmoid(gf);
            const float i  = fast_sigmoid(gi);
            const float ct = fast_tanh(gc);
            const float o  = fast_sigmoid(go);

            const size_t idx = (size_t)row * H_DIM + n;
            const float cp = c_prev[idx];
            const float cn = f * cp + i * ct;
            const float hn = o * fast_tanh(cn);
            c_out[idx] = cn;
            h_out[idx] = hn;
        }
    }
}

// ---------------------------------------------------------------------------
// launcher
// ---------------------------------------------------------------------------
extern "C" void kernel_launch(void* const* d_in, const int* in_sizes, int n_in,
                              void* d_out, int out_size, void* d_ws, size_t ws_size,
                              hipStream_t stream) {
    const float* x      = (const float*)d_in[0];
    const float* c_prev = (const float*)d_in[1];
    const float* h_prev = (const float*)d_in[2];
    const float* W_x    = (const float*)d_in[3];
    const float* W_h    = (const float*)d_in[4];
    const float* b_x    = (const float*)d_in[5];
    const float* b_h    = (const float*)d_in[6];

    float* c_out = (float*)d_out;
    float* h_out = c_out + (size_t)B_DIM * H_DIM;

    // workspace: bf16 copies of x, h_prev, W_x, W_h (128 MB total)
    bf16_t* x_bf  = (bf16_t*)d_ws;
    bf16_t* h_bf  = x_bf  + (size_t)B_DIM * I_DIM;
    bf16_t* wx_bf = h_bf  + (size_t)B_DIM * H_DIM;
    bf16_t* wh_bf = wx_bf + (size_t)4 * H_DIM * I_DIM;

    const int nx = B_DIM * I_DIM;
    const int nh = B_DIM * H_DIM;
    const int nw = 4 * H_DIM * I_DIM;

    cvt_f32_bf16_kernel<<<2048, 256, 0, stream>>>(x,      x_bf,  nx);
    cvt_f32_bf16_kernel<<<2048, 256, 0, stream>>>(h_prev, h_bf,  nh);
    cvt_f32_bf16_kernel<<<2048, 256, 0, stream>>>(W_x,    wx_bf, nw);
    cvt_f32_bf16_kernel<<<2048, 256, 0, stream>>>(W_h,    wh_bf, nw);

    dim3 grid(H_DIM / 64, B_DIM / 128);   // (32, 64)
    lstm_cell_wmma_kernel<<<grid, 256, 0, stream>>>(
        x_bf, h_bf, wx_bf, wh_bf, c_prev, b_x, b_h, c_out, h_out);
}
